// LMMDLoss_22179211117322
// MI455X (gfx1250) — matrix operand order repaired and checked by reading, not compiled, not run on hardware
//
#include <hip/hip_runtime.h>
#include <hip/hip_bf16.h>

// ---------------------------------------------------------------------------
// LMMD loss, fused for MI455X (gfx1250, wave32, WMMA).
//   B=4096 rows per domain, D=256 features, C=10 classes, N=8192 total rows.
// v2: bandwidth pass replaced by closed form (sum of pairwise L2 is
//     2N*sum(sq) - 2*||colsum||^2), SS/TT blocks use symmetry (upper
//     triangle x2 + masked diagonal tiles). ~3x less work than v1.
// f32 accuracy recovered from f16 WMMA via hi/lo split (3 WMMAs per chunk).
// ---------------------------------------------------------------------------

#define BROWS 4096
#define DFEAT 256
#define NCLS  10
#define NTOT  8192

typedef __attribute__((ext_vector_type(16))) _Float16 v16h;
typedef __attribute__((ext_vector_type(8)))  _Float16 v8h_t;
typedef __attribute__((ext_vector_type(8)))  float    v8f;

#define WMMA1(A, Bm, Cc) \
  __builtin_amdgcn_wmma_f32_16x16x32_f16(false, (A), false, (Bm), (short)0, (Cc), false, false)

#if __has_builtin(__builtin_amdgcn_exp2f)
#define EXP2F(x) __builtin_amdgcn_exp2f(x)
#else
#define EXP2F(x) exp2f(x)
#endif

// Workspace byte offsets (all 64B aligned). Total ~8.7 MB.
#define XHI_OFF  ((size_t)0)                          // N*D f16 hi parts  (4 MiB)
#define XLO_OFF  ((size_t)(NTOT * DFEAT * 2))         // N*D f16 lo parts  (4 MiB)
#define SQ_OFF   (XLO_OFF + (size_t)NTOT * DFEAT * 2) // N f32 row norms   (32 KiB)
#define SMH_OFF  (SQ_OFF + (size_t)NTOT * 4)          // B*16 f16 src wts  (128 KiB)
#define TMH_OFF  (SMH_OFF + (size_t)BROWS * 16 * 2)   // B*16 f16 tgt wts  (128 KiB)
#define COL_OFF  (TMH_OFF + (size_t)BROWS * 16 * 2)   // D f32 column sums (1 KiB)
#define SC_OFF   (COL_OFF + (size_t)DFEAT * 4)        // 64 f32 scalars
// scalar slots: [0..15] wscale_s, [16..31] wscale_t, [32] count,
//               [34] exp2 coef, [35] main accumulator

// ------------------------- fragment loaders --------------------------------
// A (16x32 f16): lane<16 -> M=lane, K=[hs*16..hs*16+15], hs=lane>>4
__device__ __forceinline__ v16h load_a(const _Float16* X, int rowbase, int k0, int lane) {
  int m = lane & 15, hs = lane >> 4;
  const _Float16* p = X + (size_t)(rowbase + m) * DFEAT + k0 + hs * 16;
  v8h_t q0 = *(const v8h_t*)p;
  v8h_t q1 = *(const v8h_t*)(p + 8);
  v16h a;
#pragma unroll
  for (int i = 0; i < 8; ++i) { a[i] = q0[i]; a[8 + i] = q1[i]; }
  return a;
}
// B (32x16 f16): lane<16 -> N=lane, K=0..15; lane>=16 -> N=lane-16, K=16..31
__device__ __forceinline__ v16h load_b(const _Float16* X, int colbase, int k0, int lane) {
  int n = lane & 15, hs = lane >> 4;
  return *(const v16h*)(X + (size_t)(colbase + n) * DFEAT + k0 + hs * 16);
}
// Weight A frag: rows of padded 16-wide weight vectors, K upper half = 0
__device__ __forceinline__ v16h load_wa(const _Float16* W, int rowbase, int lane) {
  int m = lane & 15, hs = lane >> 4;
  v16h a;
  if (hs == 0) {
    a = *(const v16h*)(W + (size_t)(rowbase + m) * 16);
  } else {
#pragma unroll
    for (int i = 0; i < 16; ++i) a[i] = (_Float16)0;
  }
  return a;
}
__device__ __forceinline__ v16h load_wb(const _Float16* W, int colbase, int lane) {
  int n = lane & 15, hs = lane >> 4;
  v16h b;
  if (hs == 0) {
    b = *(const v16h*)(W + (size_t)(colbase + n) * 16);
  } else {
#pragma unroll
    for (int i = 0; i < 16; ++i) b[i] = (_Float16)0;
  }
  return b;
}

// 32x32 Gram block (2x2 WMMA tiles), hi/lo compensated: x.y ~ hh + hl + lh
__device__ __forceinline__ void gram32(const _Float16* xhi, const _Float16* xlo,
                                       int r0, int c0, int lane,
                                       v8f& c00, v8f& c01, v8f& c10, v8f& c11) {
  c00 = v8f{}; c01 = v8f{}; c10 = v8f{}; c11 = v8f{};
  for (int k0 = 0; k0 < DFEAT; k0 += 32) {
    v16h ah0 = load_a(xhi, r0,      k0, lane);
    v16h al0 = load_a(xlo, r0,      k0, lane);
    v16h ah1 = load_a(xhi, r0 + 16, k0, lane);
    v16h al1 = load_a(xlo, r0 + 16, k0, lane);
    v16h bh0 = load_b(xhi, c0,      k0, lane);
    v16h bl0 = load_b(xlo, c0,      k0, lane);
    v16h bh1 = load_b(xhi, c0 + 16, k0, lane);
    v16h bl1 = load_b(xlo, c0 + 16, k0, lane);
    c00 = WMMA1(ah0, bh0, c00); c00 = WMMA1(ah0, bl0, c00); c00 = WMMA1(al0, bh0, c00);
    c01 = WMMA1(ah0, bh1, c01); c01 = WMMA1(ah0, bl1, c01); c01 = WMMA1(al0, bh1, c01);
    c10 = WMMA1(ah1, bh0, c10); c10 = WMMA1(ah1, bl0, c10); c10 = WMMA1(al1, bh0, c10);
    c11 = WMMA1(ah1, bh1, c11); c11 = WMMA1(ah1, bl1, c11); c11 = WMMA1(al1, bh1, c11);
  }
}

// C/D tile layout: VGPR r holds M = r + 8*(lane>>4), N = lane&15
// scale: uniform weight multiplier; diag: per-element (i<j)?2:(i==j)?1:0 mask
__device__ __forceinline__ float consume_ker(const v8f& g, const v8f& wt,
                                             const float* sq, int rglob, int cglob,
                                             int lane, float coef,
                                             float scale, bool diag) {
  float s = 0.f;
  int m0 = rglob + ((lane >> 4) << 3);
  int j  = cglob + (lane & 15);
  float sqc = sq[j];
#pragma unroll
  for (int r = 0; r < 8; ++r) {
    int i = m0 + r;
    float l2 = fmaxf(sq[i] + sqc - 2.0f * g[r], 0.0f);
    // sum_{k=0..4} exp(-l2/(bw*2^k)) = z + z^2 + z^4 + z^8 + z^16, z=exp(-l2/(16 bw))
    float z  = EXP2F(-l2 * coef);
    float z2 = z * z, z4 = z2 * z2, z8 = z4 * z4, z16 = z8 * z8;
    float el = diag ? ((i < j) ? 2.f : ((i == j) ? 1.f : 0.f)) : scale;
    s += el * wt[r] * (z + z2 + z4 + z8 + z16);
  }
  return s;
}

// ---------------------------- kernels --------------------------------------
// 1. class stats: column sums, argmax-onehot counts -> mask, scales, zero accs
__global__ void __launch_bounds__(256) k_stats(const float* __restrict__ slab,
                                               const float* __restrict__ tlog,
                                               float* __restrict__ sc,
                                               float* __restrict__ colsum) {
  __shared__ float ss[16], st[16], scnt[16];
  int t = threadIdx.x;
  colsum[t] = 0.f;                       // zero D=256 feature column sums
  if (t < 16) { ss[t] = 0.f; st[t] = 0.f; scnt[t] = 0.f; }
  __syncthreads();
  for (int i = t; i < BROWS; i += 256) {
    float mx = -1e30f; int am = 0;
#pragma unroll
    for (int c = 0; c < NCLS; ++c) {
      float sv = slab[i * NCLS + c];
      float tv = tlog[i * NCLS + c];
      atomicAdd(&ss[c], sv);
      atomicAdd(&st[c], tv);
      if (tv > mx) { mx = tv; am = c; }
    }
    atomicAdd(&scnt[am], 1.0f);
  }
  __syncthreads();
  if (t == 0) {
    float cnt = 0.f;
    for (int c = 0; c < 16; ++c) {
      float m = (c < NCLS && ss[c] > 0.f && scnt[c] > 0.f) ? 1.f : 0.f;
      cnt += m;
      float ssafe = (c < NCLS && ss[c] != 0.f) ? ss[c] : 100.f;
      float tsafe = (c < NCLS && st[c] != 0.f) ? st[c] : 100.f;
      sc[c]      = m / ssafe;
      sc[16 + c] = m / tsafe;
    }
    sc[32] = cnt;
    sc[35] = 0.f;   // main accumulator
  }
}

// 2. hi/lo f16 split of concat(source,target) + row squared norms
__global__ void __launch_bounds__(256) k_convert(const float* __restrict__ src,
                                                 const float* __restrict__ tgt,
                                                 _Float16* __restrict__ xhi,
                                                 _Float16* __restrict__ xlo,
                                                 float* __restrict__ sq) {
  int row = blockIdx.x, d = threadIdx.x;
  const float* base = (row < BROWS) ? (src + (size_t)row * DFEAT)
                                    : (tgt + (size_t)(row - BROWS) * DFEAT);
  float x = base[d];
  _Float16 h = (_Float16)x;
  _Float16 l = (_Float16)(x - (float)h);
  xhi[(size_t)row * DFEAT + d] = h;
  xlo[(size_t)row * DFEAT + d] = l;
  __shared__ float red[256];
  red[d] = x * x;
  __syncthreads();
  for (int s = 128; s > 0; s >>= 1) { if (d < s) red[d] += red[d + s]; __syncthreads(); }
  if (d == 0) sq[row] = red[0];
}

// 3. feature column sums over all N rows (for closed-form bandwidth)
__global__ void __launch_bounds__(256) k_colsum(const float* __restrict__ src,
                                                const float* __restrict__ tgt,
                                                float* __restrict__ colsum) {
  int t = threadIdx.x;                       // feature index (D == 256)
  int rb = blockIdx.x * 128;                 // 64 blocks x 128 rows
  float s = 0.f;
  for (int r = 0; r < 128; ++r) {
    int row = rb + r;
    const float* base = (row < BROWS) ? (src + (size_t)row * DFEAT)
                                      : (tgt + (size_t)(row - BROWS) * DFEAT);
    s += base[t];
  }
  atomicAdd(&colsum[t], s);
}

// 4. padded 16-wide f16 weight rows for source / target
__global__ void __launch_bounds__(256) k_weights(const float* __restrict__ slab,
                                                 const float* __restrict__ tlog,
                                                 const float* __restrict__ sc,
                                                 _Float16* __restrict__ smh,
                                                 _Float16* __restrict__ tmh) {
  int idx = blockIdx.x * 256 + threadIdx.x;   // BROWS*16 total
  int row = idx >> 4, c = idx & 15;
  float sv = (c < NCLS) ? slab[row * NCLS + c] * sc[c]      : 0.f;
  float tv = (c < NCLS) ? tlog[row * NCLS + c] * sc[16 + c] : 0.f;
  smh[idx] = (_Float16)sv;
  tmh[idx] = (_Float16)tv;
}

// 5. closed-form bandwidth: sum(l2) = 2N*sum(sq) - 2*||colsum||^2
__global__ void __launch_bounds__(256) k_finbw(const float* __restrict__ sq,
                                               const float* __restrict__ colsum,
                                               float* __restrict__ sc) {
  __shared__ float red[256];
  int t = threadIdx.x;
  float sp = 0.f;
  for (int i = t; i < NTOT; i += 256) sp += sq[i];
  float c = colsum[t];
  red[t] = 2.0f * (float)NTOT * sp - 2.0f * c * c;
  __syncthreads();
  for (int s = 128; s > 0; s >>= 1) { if (t < s) red[t] += red[t + s]; __syncthreads(); }
  if (t == 0) {
    const float nf = (float)NTOT;
    float bw = red[0] / (nf * nf - nf);
    bw *= 0.25f;                                  // / kernel_mul^(kernel_num//2) = 4
    sc[34] = 1.4426950408889634f / (16.0f * bw);  // log2(e) / (16*bw)
  }
}

// 6. main pass: weighted multi-scale kernel sum over SS / TT / ST blocks.
//    SS/TT are symmetric: lower-triangle tiles skipped, upper x2, diag masked.
__global__ void __launch_bounds__(256) k_main(const _Float16* __restrict__ xhi,
                                              const _Float16* __restrict__ xlo,
                                              const float* __restrict__ sq,
                                              const _Float16* __restrict__ smh,
                                              const _Float16* __restrict__ tmh,
                                              float* __restrict__ sc) {
  int lane = threadIdx.x & 31;
  int w = blockIdx.x * 8 + (threadIdx.x >> 5);     // 16384 wave-tiles
  int li0 = (w >> 7) * 32, lj0 = (w & 127) * 32;   // local (0..4095) tile bases
  int zz = blockIdx.z;                             // 0=SS 1=TT 2=ST
  bool sym = (zz != 2);
  if (sym && li0 > lj0) return;                    // strictly-below-diagonal tile
  int xbase = (zz == 1) ? BROWS : 0;
  int ybase = (zz == 0) ? 0 : BROWS;
  const _Float16* WA = (zz == 1) ? tmh : smh;
  const _Float16* WB = (zz == 0) ? smh : tmh;
  float sign = (zz == 2) ? -2.0f : 1.0f;
  int r0 = xbase + li0, c0 = ybase + lj0;

  v8f g00, g01, g10, g11;
  gram32(xhi, xlo, r0, c0, lane, g00, g01, g10, g11);

  // rank-10 weight outer-product tiles via WMMA (padded to K=32)
  v16h wa0 = load_wa(WA, li0, lane), wa1 = load_wa(WA, li0 + 16, lane);
  v16h wb0 = load_wb(WB, lj0, lane), wb1 = load_wb(WB, lj0 + 16, lane);
  v8f zc = {};
  v8f w00 = WMMA1(wa0, wb0, zc);
  v8f w01 = WMMA1(wa0, wb1, zc);
  v8f w10 = WMMA1(wa1, wb0, zc);
  v8f w11 = WMMA1(wa1, wb1, zc);

  float coef = sc[34];
  float base = sym ? 2.0f : 1.0f;
  float s = 0.f;
  auto sub = [&](const v8f& g, const v8f& wt, int sr, int scl, int gr, int gc) {
    if (sym && sr > scl) return 0.f;               // below diagonal subtile
    bool dg = sym && (sr == scl);
    return consume_ker(g, wt, sq, gr, gc, lane, coef, base, dg);
  };
  s += sub(g00, w00, li0,      lj0,      r0,      c0);
  s += sub(g01, w01, li0,      lj0 + 16, r0,      c0 + 16);
  s += sub(g10, w10, li0 + 16, lj0,      r0 + 16, c0);
  s += sub(g11, w11, li0 + 16, lj0 + 16, r0 + 16, c0 + 16);
  s *= sign;
#pragma unroll
  for (int off = 16; off > 0; off >>= 1) s += __shfl_down(s, off, 32);
  if (lane == 0) atomicAdd(&sc[35], s);
}

// 7. final: divide by class count
__global__ void k_fin(const float* __restrict__ sc, float* __restrict__ out) {
  float cnt = sc[32];
  out[0] = (cnt > 0.f) ? sc[35] / cnt : 0.f;
}

// ---------------------------------------------------------------------------
extern "C" void kernel_launch(void* const* d_in, const int* in_sizes, int n_in,
                              void* d_out, int out_size, void* d_ws, size_t ws_size,
                              hipStream_t stream) {
  const float* src  = (const float*)d_in[0];   // (4096,256)
  const float* tgt  = (const float*)d_in[1];   // (4096,256)
  const float* slab = (const float*)d_in[2];   // (4096,10) one-hot
  const float* tlog = (const float*)d_in[3];   // (4096,10)

  char* ws = (char*)d_ws;
  _Float16* xhi = (_Float16*)(ws + XHI_OFF);
  _Float16* xlo = (_Float16*)(ws + XLO_OFF);
  float*    sq  = (float*)(ws + SQ_OFF);
  _Float16* smh = (_Float16*)(ws + SMH_OFF);
  _Float16* tmh = (_Float16*)(ws + TMH_OFF);
  float*    col = (float*)(ws + COL_OFF);
  float*    sc  = (float*)(ws + SC_OFF);

  k_stats  <<<1, 256, 0, stream>>>(slab, tlog, sc, col);
  k_convert<<<NTOT, 256, 0, stream>>>(src, tgt, xhi, xlo, sq);
  k_colsum <<<NTOT / 128, 256, 0, stream>>>(src, tgt, col);
  k_weights<<<(BROWS * 16) / 256, 256, 0, stream>>>(slab, tlog, sc, smh, tmh);
  k_finbw  <<<1, 256, 0, stream>>>(sq, col, sc);
  k_main   <<<dim3((BROWS / 32) * (BROWS / 32) / 8, 1, 3), 256, 0, stream>>>(
               xhi, xlo, sq, smh, tmh, sc);
  k_fin    <<<1, 1, 0, stream>>>(sc, (float*)d_out);
}
